// Qwen3_5GatedDeltaNet_51196010168554
// MI455X (gfx1250) — compile-verified, matrix operand order checked by reading, not run on hardware
//
#include <hip/hip_runtime.h>
#include <hip/hip_bf16.h>
#include <math.h>

#define DEVI __device__ __forceinline__

typedef __attribute__((ext_vector_type(16))) __bf16 bf16x16;
typedef __attribute__((ext_vector_type(8)))  float  floatx8;

// ---------------- problem constants ----------------
static constexpr int Bb = 2, Ls = 2048, HID = 2048;
static constexpr int HK = 16, HV = 32, DK = 128, DV = 128, KC = 4;
static constexpr int KEY_DIM  = HK * DK;              // 2048
static constexpr int VAL_DIM  = HV * DV;              // 4096
static constexpr int CONV_DIM = 2 * KEY_DIM + VAL_DIM;// 8192
static constexpr int CHUNK = 64;
static constexpr int NC = Ls / CHUNK;                 // 32
static constexpr int ML = Bb * Ls;                    // 4096

// ---------------- bf16 helpers (bit-level, RNE) ----------------
DEVI unsigned short f2bfb(float f) {
  unsigned u = __builtin_bit_cast(unsigned, f);
  unsigned r = u + 0x7FFFu + ((u >> 16) & 1u);
  return (unsigned short)(r >> 16);
}
DEVI float bf2f(unsigned short b) {
  return __builtin_bit_cast(float, (unsigned)b << 16);
}

union FragU { unsigned short us[16]; bf16x16 v; };

DEVI float ld_f(const float* p) { return *p; }
DEVI float ld_f(const unsigned short* p) { return bf2f(*p); }
DEVI unsigned short ld_bf(const float* p) { return f2bfb(*p); }
DEVI unsigned short ld_bf(const unsigned short* p) { return *p; }
DEVI void st_c(float* p, float v) { *p = v; }
DEVI void st_c(unsigned short* p, float v) { *p = f2bfb(v); }

DEVI floatx8 wmma_bf16(bf16x16 a, bf16x16 b, floatx8 c) {
  return __builtin_amdgcn_wmma_f32_16x16x32_bf16(false, a, false, b, (short)0, c,
                                                 false, false);
}

// ---- CDNA5 async memory->LDS (ASYNCcnt) ----
// Low 32 bits of a flat LDS pointer are the LDS byte address (ISA 10.2 aperture rules).
DEVI void async_ld_b128(const unsigned short* gsrc, unsigned short* lds_dst) {
  unsigned l32 = (unsigned)(uintptr_t)(void*)lds_dst;
  unsigned long long g = (unsigned long long)(uintptr_t)(const void*)gsrc;
  asm volatile("global_load_async_to_lds_b128 %0, %1, off"
               :: "v"(l32), "v"(g) : "memory");
}
DEVI void wait_asynccnt0() { asm volatile("s_wait_asynccnt 0x0" ::: "memory"); }

// A fragment: 16x32 (MxK). T => element A(m,k) = buf[k*ld + m]
template<bool T, typename AT>
DEVI bf16x16 load_afrag(const AT* buf, int ld, int m0, int k0, int lane) {
  const int r  = m0 + (lane & 15);
  const int kb = k0 + ((lane & 16) ? 8 : 0);
  FragU fu;
#pragma unroll
  for (int h = 0; h < 16; h++) {
    const int v = h >> 1, hh = h & 1;
    const int k = kb + 2 * v + ((v >= 4) ? 8 : 0) + hh;
    fu.us[h] = T ? ld_bf(&buf[(size_t)k * ld + r]) : ld_bf(&buf[(size_t)r * ld + k]);
  }
  return fu.v;
}
// B fragment: 32x16 (KxN). T => operand B(k,n) = buf[n*ld + k] (i.e. B = stored^T)
template<bool T, typename BT>
DEVI bf16x16 load_bfrag(const BT* buf, int ld, int k0, int n0, int lane) {
  const int n  = n0 + (lane & 15);
  const int kb = k0 + ((lane & 16) ? 8 : 0);
  FragU fu;
#pragma unroll
  for (int h = 0; h < 16; h++) {
    const int v = h >> 1, hh = h & 1;
    const int k = kb + 2 * v + ((v >= 4) ? 8 : 0) + hh;
    fu.us[h] = T ? ld_bf(&buf[(size_t)n * ld + k]) : ld_bf(&buf[(size_t)k * ld + n]);
  }
  return fu.v;
}

// C[MxN] = (Ci ? Ci : 0) + op(A) @ op(B); 8 waves cooperate; fp32 accumulate.
template<int M, int N, int K, bool TA, bool TB,
         typename COT, typename CIT, typename AT, typename BT>
DEVI void mm_tile(COT* Cd, int ldc, const CIT* Ci, int ldci,
                  const AT* A, int lda, const BT* B, int ldb, int wave, int lane) {
  constexpr int TN = N / 16, NT = (M / 16) * TN;
  const int r16 = lane & 15;
  const int h8  = (lane & 16) ? 8 : 0;
  for (int t = wave; t < NT; t += 8) {
    const int m0 = (t / TN) * 16, n0 = (t % TN) * 16;
    floatx8 acc;
    if (Ci) {
#pragma unroll
      for (int e = 0; e < 8; e++) acc[e] = ld_f(&Ci[(size_t)(m0 + h8 + e) * ldci + n0 + r16]);
    } else {
#pragma unroll
      for (int e = 0; e < 8; e++) acc[e] = 0.f;
    }
#pragma unroll
    for (int k0 = 0; k0 < K; k0 += 32)
      acc = wmma_bf16(load_afrag<TA>(A, lda, m0, k0, lane),
                      load_bfrag<TB>(B, ldb, k0, n0, lane), acc);
#pragma unroll
    for (int e = 0; e < 8; e++) st_c(&Cd[(size_t)(m0 + h8 + e) * ldc + n0 + r16], acc[e]);
  }
}

// ---------------- elementwise kernels ----------------
__global__ void cast_f32_bf16(const float* __restrict__ x,
                              unsigned short* __restrict__ y, size_t n) {
  size_t i = (size_t)blockIdx.x * blockDim.x + threadIdx.x;
  if (i < n) y[i] = f2bfb(x[i]);
}

// depthwise causal conv (K=4) + SiLU. mixed: [ML][CONV_DIM]
__global__ __launch_bounds__(256)
void conv_silu_kernel(const float* __restrict__ mixed, const float* __restrict__ cw,
                      float* __restrict__ out) {
  size_t i = (size_t)blockIdx.x * 256 + threadIdx.x;
  if (i >= (size_t)ML * CONV_DIM) return;
  int c = (int)(i % CONV_DIM);
  size_t row = i / CONV_DIM;            // b*Ls + t
  int t = (int)(row % Ls);
  float acc = 0.f;
#pragma unroll
  for (int j = 0; j < KC; j++) {
    int tt = t - (KC - 1) + j;
    if (tt >= 0) acc += cw[c * KC + j] * mixed[(row - (size_t)(t - tt)) * CONV_DIM + c];
  }
  out[i] = acc / (1.f + __expf(-acc));  // silu
}

// g = -exp(A_log)*softplus(a+dt_bias); beta = sigmoid(braw)
__global__ void gate_prep_kernel(const float* __restrict__ araw, const float* __restrict__ braw,
                                 const float* __restrict__ A_log, const float* __restrict__ dtb,
                                 float* __restrict__ g, float* __restrict__ beta) {
  int i = blockIdx.x * 256 + threadIdx.x;
  if (i >= ML * HV) return;
  int h = i % HV;
  float a = araw[i] + dtb[h];
  float sp = (a > 20.f) ? a : log1pf(__expf(a));
  g[i] = -__expf(A_log[h]) * sp;
  beta[i] = 1.f / (1.f + __expf(-braw[i]));
}

// RMS-norm over DV, * norm_w, * silu(z), write bf16 for final GEMM
__global__ __launch_bounds__(256)
void gate_norm_kernel(const float* __restrict__ core, const float* __restrict__ z,
                      const float* __restrict__ nw, unsigned short* __restrict__ gated) {
  int wid = blockIdx.x * 8 + (threadIdx.x >> 5);
  int lane = threadIdx.x & 31;
  if (wid >= ML * HV) return;
  int rowml = wid / HV, h = wid % HV;
  const float* x  = core + (size_t)wid * DV;
  const float* zr = z + (size_t)rowml * VAL_DIM + h * DV;
  float xv[4], s = 0.f;
#pragma unroll
  for (int e = 0; e < 4; e++) { xv[e] = x[lane + e * 32]; s += xv[e] * xv[e]; }
#pragma unroll
  for (int off = 16; off > 0; off >>= 1) s += __shfl_xor(s, off, 32);
  float inv = rsqrtf(s * (1.f / DV) + 1e-6f);
#pragma unroll
  for (int e = 0; e < 4; e++) {
    int d = lane + e * 32;
    float zz = zr[d];
    float val = xv[e] * inv * nw[d] * (zz / (1.f + __expf(-zz)));
    gated[(size_t)rowml * VAL_DIM + h * DV + d] = f2bfb(val);
  }
}

// ---------------- tiled bf16 GEMM: C[MxN] = A[MxK] @ W[NxK]^T ----------------
// Async double-buffered LDS staging via GLOBAL_LOAD_ASYNC_TO_LDS_B128 (ASYNCcnt).
__global__ __launch_bounds__(256)
void gemm_bf16_nt(const unsigned short* __restrict__ A, const unsigned short* __restrict__ W,
                  float* __restrict__ C, int M, int N, int K) {
  __shared__ __align__(16) unsigned short sA[2][128 * 32];
  __shared__ __align__(16) unsigned short sW[2][128 * 32];
  const int tid = threadIdx.x, lane = tid & 31, wave = tid >> 5;
  const int row0 = blockIdx.x * 128, col0 = blockIdx.y * 128;
  const int wm = wave >> 2, wn = wave & 3;   // 2x4 wave grid -> 64x32 per wave
  floatx8 acc[4][2];
#pragma unroll
  for (int i = 0; i < 4; i++)
#pragma unroll
    for (int j = 0; j < 2; j++)
#pragma unroll
      for (int e = 0; e < 8; e++) acc[i][j][e] = 0.f;

  const int ldr = tid >> 1;            // 0..127
  const int ldc16 = (tid & 1) * 16;    // 0 / 16
  const bool wok = (col0 + ldr) < N;

  // out-of-range W rows: zero once; async loads never touch them
  if (!wok) {
    uint4 z = make_uint4(0, 0, 0, 0);
    *(uint4*)(&sW[0][ldr * 32 + ldc16])     = z;
    *(uint4*)(&sW[0][ldr * 32 + ldc16 + 8]) = z;
    *(uint4*)(&sW[1][ldr * 32 + ldc16])     = z;
    *(uint4*)(&sW[1][ldr * 32 + ldc16 + 8]) = z;
  }
  const unsigned short* aSrc = A + (size_t)(row0 + ldr) * K + ldc16;
  const unsigned short* wSrc = W + (size_t)(col0 + ldr) * K + ldc16;

  // stage K-slice kt into LDS buffer buf (16B per async op, per-lane LDS dest)
  auto stage = [&](int buf, int kt) {
    async_ld_b128(aSrc + kt,     &sA[buf][ldr * 32 + ldc16]);
    async_ld_b128(aSrc + kt + 8, &sA[buf][ldr * 32 + ldc16 + 8]);
    if (wok) {
      async_ld_b128(wSrc + kt,     &sW[buf][ldr * 32 + ldc16]);
      async_ld_b128(wSrc + kt + 8, &sW[buf][ldr * 32 + ldc16 + 8]);
    }
  };

  stage(0, 0);
  wait_asynccnt0();
  __syncthreads();

  for (int kt = 0; kt < K; kt += 32) {
    const int cur = (kt >> 5) & 1;
    if (kt + 32 < K) {
      stage(cur ^ 1, kt + 32);
      __builtin_prefetch((const void*)(aSrc + kt + 64), 0, 1);
    }
    bf16x16 bfr[2];
#pragma unroll
    for (int tn = 0; tn < 2; tn++)
      bfr[tn] = load_bfrag<true>(sW[cur], 32, 0, wn * 32 + tn * 16, lane);
#pragma unroll
    for (int tm = 0; tm < 4; tm++) {
      bf16x16 af = load_afrag<false>(sA[cur], 32, wm * 64 + tm * 16, 0, lane);
#pragma unroll
      for (int tn = 0; tn < 2; tn++) acc[tm][tn] = wmma_bf16(af, bfr[tn], acc[tm][tn]);
    }
    wait_asynccnt0();   // this wave's next-tile async loads have landed in LDS
    __syncthreads();    // all waves done reading cur + all async writes visible
  }

  const int r16 = lane & 15, h8 = (lane & 16) ? 8 : 0;
#pragma unroll
  for (int tm = 0; tm < 4; tm++)
#pragma unroll
    for (int tn = 0; tn < 2; tn++) {
      int col = col0 + wn * 32 + tn * 16 + r16;
      if (col < N) {
#pragma unroll
        for (int e = 0; e < 8; e++)
          C[(size_t)(row0 + wm * 64 + tm * 16 + h8 + e) * N + col] = acc[tm][tn][e];
      }
    }
}

// ---------------- chunked gated delta rule: one block per (b, hv) ----------------
static constexpr size_t DSM_S   = 128 * 128 * sizeof(float);
static constexpr size_t DSM_BUF = 64 * 128 * sizeof(unsigned short);
static constexpr size_t DSM_M   = 64 * 64 * sizeof(float);
static constexpr size_t DELTA_SMEM =
    DSM_S + 5 * DSM_BUF + 3 * DSM_M + 256 * sizeof(float) + 5 * 64 * sizeof(float);

__global__ __launch_bounds__(256)
void delta_kernel(const float* __restrict__ qkv, const float* __restrict__ g,
                  const float* __restrict__ beta, float* __restrict__ core) {
  extern __shared__ char smem[];
  char* p = smem;
  float* S = (float*)p;            p += DSM_S;
  unsigned short* qs = (unsigned short*)p; p += DSM_BUF;
  unsigned short* ks = (unsigned short*)p; p += DSM_BUF;
  unsigned short* vs = (unsigned short*)p; p += DSM_BUF;
  unsigned short* us = (unsigned short*)p; p += DSM_BUF;
  unsigned short* ws = (unsigned short*)p; p += DSM_BUF;
  float* Matn = (float*)p;         p += DSM_M;    // attn
  float* Ma   = (float*)p;         p += DSM_M;    // kb@k^T -> A
  float* Mdec = (float*)p;         p += DSM_M;    // decay -> T
  float* red  = (float*)p;         p += 256 * sizeof(float);
  float* gcs  = (float*)p;         p += 64 * sizeof(float);
  float* qn   = (float*)p;         p += 64 * sizeof(float);
  float* kn   = (float*)p;         p += 64 * sizeof(float);
  float* brow = (float*)p;         p += 64 * sizeof(float);
  float* grow = (float*)p;         p += 64 * sizeof(float);

  const int blk = blockIdx.x, b = blk / HV, hv = blk % HV, qh = hv >> 1;
  const int tid = threadIdx.x, lane = tid & 31, wave = tid >> 5;
  const int r = tid >> 2, sg = tid & 3;

  for (int i = tid; i < 128 * 128; i += 256) S[i] = 0.f;
  __syncthreads();

  for (int ci = 0; ci < NC; ci++) {
    const int l0 = ci * CHUNK;
    const size_t rowg = (size_t)(b * Ls + l0 + r) * CONV_DIM;
    const float* qrow = qkv + rowg + qh * DK;
    const float* krow = qkv + rowg + KEY_DIM + qh * DK;
    const float* vrow = qkv + rowg + 2 * KEY_DIM + hv * DV;

    // q pass: raw -> bf16, sum of squares
    float s = 0.f;
#pragma unroll
    for (int e = 0; e < 32; e++) {
      float x = qrow[sg * 32 + e]; s += x * x; qs[r * 128 + sg * 32 + e] = f2bfb(x);
    }
    red[tid] = s;
    __syncthreads();
    if (sg == 0)
      qn[r] = rsqrtf(red[r*4] + red[r*4+1] + red[r*4+2] + red[r*4+3] + 1e-6f) * 0.08838834764831845f;
    __syncthreads();
    // k pass
    s = 0.f;
#pragma unroll
    for (int e = 0; e < 32; e++) {
      float x = krow[sg * 32 + e]; s += x * x; ks[r * 128 + sg * 32 + e] = f2bfb(x);
    }
    red[tid] = s;
    if (sg == 0) {
      int gi = (b * Ls + l0 + r) * HV + hv;
      brow[r] = beta[gi]; grow[r] = g[gi];
    }
    __syncthreads();
    if (sg == 0) kn[r] = rsqrtf(red[r*4] + red[r*4+1] + red[r*4+2] + red[r*4+3] + 1e-6f);
    if (tid == 0) { float a = 0.f; for (int t = 0; t < 64; t++) { a += grow[t]; gcs[t] = a; } }
    __syncthreads();
    // rescale q,k; build kb (us) and vb (vs)
#pragma unroll
    for (int e = 0; e < 32; e++) {
      int idx = r * 128 + sg * 32 + e;
      float qv = bf2f(qs[idx]) * qn[r]; qs[idx] = f2bfb(qv);
      float kv = bf2f(ks[idx]) * kn[r]; ks[idx] = f2bfb(kv);
      us[idx] = f2bfb(kv * brow[r]);
      vs[idx] = f2bfb(vrow[sg * 32 + e] * brow[r]);
    }
    __syncthreads();
    // decay matrix
    for (int idx = tid; idx < 64 * 64; idx += 256) {
      int i = idx >> 6, j = idx & 63;
      Mdec[idx] = (j <= i) ? __expf(gcs[i] - gcs[j]) : 0.f;
    }
    __syncthreads();
    // A = -tril(kb @ k^T * decay, -1)
    mm_tile<64, 64, 128, false, true>(Ma, 64, (const float*)nullptr, 0, us, 128, ks, 128, wave, lane);
    __syncthreads();
    for (int idx = tid; idx < 64 * 64; idx += 256) {
      int i = idx >> 6, j = idx & 63;
      Ma[idx] = (j < i) ? -Ma[idx] * Mdec[idx] : 0.f;
    }
    __syncthreads();
    // attn = (q @ k^T) * decay
    mm_tile<64, 64, 128, false, true>(Matn, 64, (const float*)nullptr, 0, qs, 128, ks, 128, wave, lane);
    __syncthreads();
    for (int idx = tid; idx < 64 * 64; idx += 256) {
      int i = idx >> 6, j = idx & 63;
      Matn[idx] = (j <= i) ? Matn[idx] * Mdec[idx] : 0.f;
    }
    __syncthreads();
    // T = (I - A)^-1 via forward substitution (unit lower), into Mdec
    if (tid < 64) Mdec[tid] = (tid == 0) ? 1.f : 0.f;
    __syncthreads();
    for (int i = 1; i < 64; i++) {
      if (tid < 64) {
        int j = tid;
        float t = 0.f;
        if (j < i) { for (int q = j; q < i; q++) t += Ma[i * 64 + q] * Mdec[q * 64 + j]; }
        else if (j == i) t = 1.f;
        Mdec[i * 64 + j] = t;
      }
      __syncthreads();
    }
    // u = T @ vb  (into us; kb no longer needed)
    mm_tile<64, 128, 64, false, false>(us, 128, (const float*)nullptr, 0, Mdec, 64, vs, 128, wave, lane);
    __syncthreads();
    // vs <- -kg = -(k * beta * exp(gcs))
    for (int idx = tid; idx < 64 * 128; idx += 256) {
      int t = idx >> 7;
      vs[idx] = f2bfb(-bf2f(ks[idx]) * brow[t] * __expf(gcs[t]));
    }
    __syncthreads();
    // ws = T @ (-kg) = -w
    mm_tile<64, 128, 64, false, false>(ws, 128, (const float*)nullptr, 0, Mdec, 64, vs, 128, wave, lane);
    __syncthreads();
    // vs <- v_new = u + (-w) @ S
    mm_tile<64, 128, 128, false, false>(vs, 128, us, 128, ws, 128, S, 128, wave, lane);
    __syncthreads();
    // qs <- q * exp(gcs)
    for (int idx = tid; idx < 64 * 128; idx += 256) {
      int t = idx >> 7;
      qs[idx] = f2bfb(bf2f(qs[idx]) * __expf(gcs[t]));
    }
    __syncthreads();
    // out = qg @ S + attn @ v_new  (fused, straight to global)
    {
      float* outp = core + ((size_t)(b * Ls + l0) * HV + hv) * DV;
      const int r16 = lane & 15, h8 = (lane & 16) ? 8 : 0;
      for (int t = wave; t < 32; t += 8) {
        int m0 = (t >> 3) * 16, n0 = (t & 7) * 16;
        floatx8 acc;
#pragma unroll
        for (int e = 0; e < 8; e++) acc[e] = 0.f;
#pragma unroll
        for (int k0 = 0; k0 < 128; k0 += 32)
          acc = wmma_bf16(load_afrag<false>(qs, 128, m0, k0, lane),
                          load_bfrag<false>(S, 128, k0, n0, lane), acc);
#pragma unroll
        for (int k0 = 0; k0 < 64; k0 += 32)
          acc = wmma_bf16(load_afrag<false>(Matn, 64, m0, k0, lane),
                          load_bfrag<false>(vs, 128, k0, n0, lane), acc);
#pragma unroll
        for (int e = 0; e < 8; e++)
          outp[(size_t)(m0 + h8 + e) * (HV * DV) + n0 + r16] = acc[e];
      }
    }
    __syncthreads();
    // state update: S = S*exp(g_last) + (k*exp(g_last-gcs))^T @ v_new
    const float glast = gcs[63];
    const float egl = __expf(glast);
    for (int idx = tid; idx < 128 * 128; idx += 256) S[idx] *= egl;
    for (int idx = tid; idx < 64 * 128; idx += 256) {
      int t = idx >> 7;
      ks[idx] = f2bfb(bf2f(ks[idx]) * __expf(glast - gcs[t]));
    }
    __syncthreads();
    mm_tile<128, 128, 64, true, false>(S, 128, (const float*)S, 128, ks, 128, vs, 128, wave, lane);
    __syncthreads();
  }
}

// ---------------- host launcher ----------------
extern "C" void kernel_launch(void* const* d_in, const int* in_sizes, int n_in,
                              void* d_out, int out_size, void* d_ws, size_t ws_size,
                              hipStream_t stream) {
  (void)in_sizes; (void)n_in; (void)out_size; (void)ws_size;
  const float* hs    = (const float*)d_in[0];
  const float* Wqkv  = (const float*)d_in[1];
  const float* convw = (const float*)d_in[2];
  const float* Wz    = (const float*)d_in[3];
  const float* Wb    = (const float*)d_in[4];
  const float* Wa    = (const float*)d_in[5];
  const float* dtb   = (const float*)d_in[6];
  const float* Alog  = (const float*)d_in[7];
  const float* normw = (const float*)d_in[8];
  const float* Wout  = (const float*)d_in[9];
  float* out = (float*)d_out;

  char* ws = (char*)d_ws;
  auto alloc = [&](size_t bytes) { char* q = ws; ws += (bytes + 255) & ~(size_t)255; return q; };
  unsigned short* hbf   = (unsigned short*)alloc((size_t)ML * HID * 2);
  unsigned short* wqkvb = (unsigned short*)alloc((size_t)CONV_DIM * HID * 2);
  unsigned short* wzb   = (unsigned short*)alloc((size_t)VAL_DIM * HID * 2);
  unsigned short* wbb   = (unsigned short*)alloc((size_t)HV * HID * 2);
  unsigned short* wab   = (unsigned short*)alloc((size_t)HV * HID * 2);
  unsigned short* woutb = (unsigned short*)alloc((size_t)HID * VAL_DIM * 2);
  float* mixed = (float*)alloc((size_t)ML * CONV_DIM * 4);  // reused as z + core later
  float* qkvs  = (float*)alloc((size_t)ML * CONV_DIM * 4);
  float* braw  = (float*)alloc((size_t)ML * HV * 4);
  float* araw  = (float*)alloc((size_t)ML * HV * 4);
  float* gbuf  = (float*)alloc((size_t)ML * HV * 4);
  float* bbuf  = (float*)alloc((size_t)ML * HV * 4);
  unsigned short* gated = (unsigned short*)alloc((size_t)ML * VAL_DIM * 2);
  float* zbuf = mixed;                              // after conv consumes mixed
  float* core = mixed + (size_t)ML * VAL_DIM;

  auto cast = [&](const float* src, unsigned short* dst, size_t n) {
    cast_f32_bf16<<<dim3((unsigned)((n + 255) / 256)), dim3(256), 0, stream>>>(src, dst, n);
  };
  cast(hs, hbf, (size_t)ML * HID);
  cast(Wqkv, wqkvb, (size_t)CONV_DIM * HID);
  cast(Wz, wzb, (size_t)VAL_DIM * HID);
  cast(Wb, wbb, (size_t)HV * HID);
  cast(Wa, wab, (size_t)HV * HID);
  cast(Wout, woutb, (size_t)HID * VAL_DIM);

  // mixed = hidden @ W_qkv^T
  gemm_bf16_nt<<<dim3(ML / 128, CONV_DIM / 128), 256, 0, stream>>>(hbf, wqkvb, mixed, ML, CONV_DIM, HID);
  // depthwise conv + silu
  conv_silu_kernel<<<dim3((unsigned)(((size_t)ML * CONV_DIM) / 256)), 256, 0, stream>>>(mixed, convw, qkvs);
  // z / beta / a projections (z overwrites mixed region — conv already consumed it)
  gemm_bf16_nt<<<dim3(ML / 128, VAL_DIM / 128), 256, 0, stream>>>(hbf, wzb, zbuf, ML, VAL_DIM, HID);
  gemm_bf16_nt<<<dim3(ML / 128, 1), 256, 0, stream>>>(hbf, wbb, braw, ML, HV, HID);
  gemm_bf16_nt<<<dim3(ML / 128, 1), 256, 0, stream>>>(hbf, wab, araw, ML, HV, HID);
  gate_prep_kernel<<<dim3((ML * HV) / 256), 256, 0, stream>>>(araw, braw, Alog, dtb, gbuf, bbuf);
  // recurrent core: 64 blocks, state S resident in LDS
  delta_kernel<<<dim3(Bb * HV), 256, DELTA_SMEM, stream>>>(qkvs, gbuf, bbuf, core);
  // rmsnorm * norm_w * silu(z) -> bf16
  gate_norm_kernel<<<dim3((ML * HV) / 8), 256, 0, stream>>>(core, zbuf, normw, gated);
  // final projection
  gemm_bf16_nt<<<dim3(ML / 128, HID / 128), 256, 0, stream>>>(gated, woutb, out, ML, HID, VAL_DIM);
}
